// BlockCirculant_50225347559618
// MI455X (gfx1250) — matrix operand order repaired and checked by reading, not compiled, not run on hardware
//
#include <hip/hip_runtime.h>
#include <stdint.h>

// Problem geometry
#define BATCH 4096
#define DIN   4096
#define DOUT  4096
#define BSZ   128
#define KIN   (DIN / BSZ)     // 32
#define KOUT  (DOUT / BSZ)    // 32

// WMMA fragment geometry (v_wmma_f32_16x16x32_bf16)
#define FM 16
#define FN 16
#define FK 32
#define MT (BATCH / FM)       // 256 M-tiles
#define NT (DOUT / FN)        // 256 N-tiles
#define KT (DIN / FK)         // 128 K-steps

// GEMM tiling: block 128x128, 8 waves, wave tile 32x64, K-chunk 64
#define BM 128
#define BN 128
#define BK 64
#define NCHUNK (DIN / BK)     // 64

typedef __attribute__((ext_vector_type(16))) __bf16 v16bf;
typedef __attribute__((ext_vector_type(8)))  float  v8f;
typedef __attribute__((ext_vector_type(4)))  int    v4i;

// ---------------------------------------------------------------------------
// Async global->LDS copy (GLOBAL_LOAD_ASYNC_TO_LDS_B128, ASYNCcnt).
// Signature (probe-confirmed): (v4i as(1)*, v4i as(3)*, imm offset, imm cpol).
// ---------------------------------------------------------------------------
#if __has_builtin(__builtin_amdgcn_global_load_async_to_lds_b128)
#define USE_ASYNC_COPY 1
typedef __attribute__((address_space(1))) v4i* as1_v4i;
typedef __attribute__((address_space(3))) v4i* as3_v4i;
#else
#define USE_ASYNC_COPY 0
#endif

__device__ __forceinline__ void cp16(__bf16* dst_lds, const __bf16* src_glb) {
#if USE_ASYNC_COPY
  __builtin_amdgcn_global_load_async_to_lds_b128(
      (as1_v4i)(void*)src_glb, (as3_v4i)(void*)dst_lds, 0, 0);
#else
  *(uint4*)dst_lds = *(const uint4*)src_glb;
#endif
}

__device__ __forceinline__ void wait_async() {
#if USE_ASYNC_COPY
#if __has_builtin(__builtin_amdgcn_s_wait_asynccnt)
  __builtin_amdgcn_s_wait_asynccnt(0);
#else
  asm volatile("s_wait_asynccnt 0" ::: "memory");
#endif
#endif
}

// ---------------------------------------------------------------------------
// Fragment interior layout (1KB = 512 bf16), bank-conflict-free:
//   elems [l*8     .. l*8+7]  = lane l, low  16B  (LDS 16B-unit l)
//   elems [256+l*8 .. +7]     = lane l, high 16B  (LDS 16B-unit 32+l)
// Each 16B ds_load covers every bank-residue mod 16 exactly twice.
// ---------------------------------------------------------------------------
__device__ __forceinline__ v16bf load_frag(const __bf16* frag, int lane) {
  union { v16bf v; uint4 q[2]; } u;
  u.q[0] = *(const uint4*)(frag + lane * 8);
  u.q[1] = *(const uint4*)(frag + 256 + lane * 8);
  return u.v;
}

// ---------------------------------------------------------------------------
// prep_x: Xp[mt][kt] fragments = bf16(x * D), A-fragment layout.
// 16-bit A 16x32 striping: lanes 0-15 hold K 0..15; lanes 16-31
// hold K 16..31 (contiguous, same order as prep_b).
// ---------------------------------------------------------------------------
__global__ void prep_x_kernel(const float* __restrict__ x,
                              const float* __restrict__ D,
                              __bf16* __restrict__ Xp) {
  int task = blockIdx.x * 256 + threadIdx.x;   // MT*KT*32 tasks
  int lane = task & 31;
  int frag = task >> 5;                        // mt*KT + kt
  int kt   = frag % KT;
  int mt   = frag / KT;
  int m    = mt * FM + (lane & 15);
  int k0   = kt * FK + (lane >> 4) * 16;
  const float* xr = x + (size_t)m * DIN;
  union { __bf16 h[16]; uint4 q[2]; } u;
#pragma unroll
  for (int qq = 0; qq < 8; ++qq) {
    int ka = k0 + qq;
    int kb = k0 + 8 + qq;
    u.h[qq]     = (__bf16)(xr[ka] * D[ka]);   // low 16B: K = khalf*16 + qq
    u.h[8 + qq] = (__bf16)(xr[kb] * D[kb]);   // high 16B: K = khalf*16 + 8 + qq
  }
  __bf16* base = Xp + (size_t)frag * 512;
  *(uint4*)(base + lane * 8)       = u.q[0];
  *(uint4*)(base + 256 + lane * 8) = u.q[1];
}

// ---------------------------------------------------------------------------
// prep_b: circulant weights in B-fragment layout.
// B[j*128+s][i*128+t] = W[i][j][(s-t) mod 128].
// 16-bit B 32x16 striping: lanes 0-15: N=lane, K 0..15; lanes 16-31: K 16..31.
// ---------------------------------------------------------------------------
__global__ void prep_b_kernel(const float* __restrict__ W,
                              __bf16* __restrict__ Bp) {
  int task = blockIdx.x * 256 + threadIdx.x;   // KT*NT*32 tasks
  int lane = task & 31;
  int frag = task >> 5;                        // kt*NT + nt
  int nt   = frag % NT;
  int kt   = frag / NT;
  int n    = nt * FN + (lane & 15);
  int i    = n >> 7;                           // output circulant block
  int t    = n & (BSZ - 1);
  int kbase = kt * FK + (lane >> 4) * 16;
  int j     = kbase >> 7;                      // input circulant block
  int sbase = kbase & (BSZ - 1);
  const float* w = W + ((size_t)i * KIN + j) * BSZ;
  union { __bf16 h[16]; uint4 q[2]; } u;
#pragma unroll
  for (int qq = 0; qq < 16; ++qq) {
    int s = sbase + qq;
    u.h[qq] = (__bf16)w[(s - t) & (BSZ - 1)];
  }
  __bf16* base = Bp + (size_t)frag * 512;
  *(uint4*)(base + lane * 8)       = u.q[0];
  *(uint4*)(base + 256 + lane * 8) = u.q[1];
}

// ---------------------------------------------------------------------------
// GEMM: C[4096x4096] = Xd * B + bias, bf16 WMMA, f32 accumulate.
// ---------------------------------------------------------------------------
__global__ void __launch_bounds__(256)
gemm_kernel(const __bf16* __restrict__ Xp, const __bf16* __restrict__ Bp,
            const float* __restrict__ bias, float* __restrict__ out) {
  // Per buffer: 2 K-steps x 8 tiles x 1KB fragments for A and B. 64KB total.
  __shared__ __align__(128) __bf16 sX[2][2][8][512];
  __shared__ __align__(128) __bf16 sB[2][2][8][512];

  const int tid  = threadIdx.x;
  const int lane = tid & 31;
  const int wid  = tid >> 5;        // 8 waves (wave32)
  const int wm   = wid & 3;         // wave M offset = wm*32 (2 M-tiles)
  const int wn   = wid >> 2;        // wave N offset = wn*64 (4 N-tiles)
  const int mb   = blockIdx.y;
  const int nb   = blockIdx.x;

  v8f acc[2][4] = {};

  auto stage = [&](int c, int buf) {
#pragma unroll
    for (int r = 0; r < 4; ++r) {
      int flat = r * 256 + tid;     // 1024 x 16B per operand = 16KB each
      int f    = flat >> 6;         // fragment 0..15
      int o    = (flat & 63) * 8;   // element offset within fragment
      int i = f & 7, j = f >> 3;
      cp16(&sX[buf][j][i][o],
           Xp + ((size_t)(mb * 8 + i) * KT + ((size_t)c * 2 + j)) * 512 + o);
      cp16(&sB[buf][j][i][o],
           Bp + (((size_t)c * 2 + j) * NT + (nb * 8 + i)) * 512 + o);
    }
  };

  stage(0, 0);
  for (int c = 0; c < NCHUNK; ++c) {
    const int buf = c & 1;
    wait_async();
    __syncthreads();                      // chunk c resident in `buf`
    if (c + 1 < NCHUNK) stage(c + 1, buf ^ 1);

    // Preload all 12 fragments, then stream 16 WMMAs back-to-back on the
    // XDL pipe (disjoint accumulators -> no D->A/B hazards).
    v16bf af[2][2];
    v16bf bq[2][4];
#pragma unroll
    for (int j = 0; j < 2; ++j) {
      af[j][0] = load_frag(&sX[buf][j][wm * 2 + 0][0], lane);
      af[j][1] = load_frag(&sX[buf][j][wm * 2 + 1][0], lane);
#pragma unroll
      for (int q = 0; q < 4; ++q)
        bq[j][q] = load_frag(&sB[buf][j][wn * 4 + q][0], lane);
    }
#pragma unroll
    for (int j = 0; j < 2; ++j) {
#pragma unroll
      for (int q = 0; q < 4; ++q) {
        acc[0][q] = __builtin_amdgcn_wmma_f32_16x16x32_bf16(
            false, af[j][0], false, bq[j][q], (short)0, acc[0][q], false, false);
        acc[1][q] = __builtin_amdgcn_wmma_f32_16x16x32_bf16(
            false, af[j][1], false, bq[j][q], (short)0, acc[1][q], false, false);
      }
    }
    __syncthreads();                      // all reads of `buf` done
  }

  // Epilogue: C/D layout — VGPR r: lanes 0-15 -> (M=r, N=lane),
  // lanes 16-31 -> (M=8+r, N=lane-16). Add bias, store f32.
  const int ncol  = lane & 15;
  const int mhalf = lane >> 4;
#pragma unroll
  for (int i = 0; i < 2; ++i) {
#pragma unroll
    for (int q = 0; q < 4; ++q) {
      int gm0 = mb * BM + wm * 32 + i * 16 + mhalf * 8;
      int gn  = nb * BN + wn * 64 + q * 16 + ncol;
      float bv = bias[gn];
      float* op = out + (size_t)gm0 * DOUT + gn;
#pragma unroll
      for (int r = 0; r < 8; ++r)
        op[(size_t)r * DOUT] = acc[i][q][r] + bv;
    }
  }
}

// ---------------------------------------------------------------------------
extern "C" void kernel_launch(void* const* d_in, const int* in_sizes, int n_in,
                              void* d_out, int out_size, void* d_ws, size_t ws_size,
                              hipStream_t stream) {
  (void)in_sizes; (void)n_in; (void)out_size; (void)ws_size;
  const float* x    = (const float*)d_in[0];
  const float* W    = (const float*)d_in[1];
  const float* D    = (const float*)d_in[2];
  const float* bias = (const float*)d_in[3];

  __bf16* Xp = (__bf16*)d_ws;                       // 32 MB packed activations
  __bf16* Bp = Xp + (size_t)BATCH * DIN;            // 32 MB packed circulant
  float*  out = (float*)d_out;

  prep_x_kernel<<<dim3((MT * KT * 32) / 256), 256, 0, stream>>>(x, D, Xp);
  prep_b_kernel<<<dim3((KT * NT * 32) / 256), 256, 0, stream>>>(W, Bp);
  gemm_kernel<<<dim3(DOUT / BN, BATCH / BM), 256, 0, stream>>>(Xp, Bp, bias, out);
}
